// RegionRouter_50122268344640
// MI455X (gfx1250) — compile-verified, matrix operand order checked by reading, not run on hardware
//
#include <hip/hip_runtime.h>
#include <hip/hip_bf16.h>

typedef __attribute__((ext_vector_type(4)))  float     v4f;
typedef __attribute__((ext_vector_type(8)))  float     v8f;
typedef __attribute__((ext_vector_type(16))) _Float16  v16h;

#define BATCH   64
#define C_IN    103
#define KPAD    128
#define HIDDEN  64
#define NREG    3
#define NEXP    3
#define HW      65536   // 256*256

// ---------------------------------------------------------------------------
// Kernel 1: global average pool over each (b,c) plane.  Bandwidth-bound:
// 1.73 GB of NT b128 streaming loads.  One block per plane.
// ---------------------------------------------------------------------------
__global__ __launch_bounds__(256) void pool_kernel(const float* __restrict__ x,
                                                   float* __restrict__ pooled) {
    const int bc  = blockIdx.x;            // plane index 0..B*C-1
    const int tid = threadIdx.x;           // 0..255
    const v4f* p  = (const v4f*)(x + (size_t)bc * HW);

    v4f acc = {0.f, 0.f, 0.f, 0.f};
    #pragma unroll 8
    for (int i = 0; i < 64; ++i) {
        v4f v = __builtin_nontemporal_load(&p[i * 256 + tid]);
        acc += v;                           // 4 independent accumulation chains
    }
    float s = acc.x + acc.y + acc.z + acc.w;

    // wave32 reduction
    #pragma unroll
    for (int off = 16; off > 0; off >>= 1)
        s += __shfl_down(s, off, 32);

    __shared__ float wsum[8];
    const int lane = tid & 31, wid = tid >> 5;
    if (lane == 0) wsum[wid] = s;
    __syncthreads();
    if (tid == 0) {
        float t = 0.f;
        #pragma unroll
        for (int i = 0; i < 8; ++i) t += wsum[i];
        pooled[bc] = t * (1.0f / (float)HW);
    }
}

// ---------------------------------------------------------------------------
// Kernel 2: layer-1 gate GEMM per region with v_wmma_f32_16x16x32_f16.
// h[b,r,n] = relu( sum_c pooled[b,c] * W1[r,n,c] + b1[r,n] )
// M=64 (batch), N=64 (hidden), K=103 padded to 128.  One workgroup, 8 waves,
// 16 tiles per region (2 per wave), 4 k-steps of 32 per tile.
// ---------------------------------------------------------------------------
__global__ __launch_bounds__(256) void gate1_wmma(const float* __restrict__ pooled,
                                                  const float* __restrict__ W1,
                                                  const float* __restrict__ b1,
                                                  float* __restrict__ hbuf) {
    __shared__ _Float16 Ah[BATCH * KPAD];    // [b][k]   16 KB
    __shared__ _Float16 Bh[KPAD * HIDDEN];   // [k][n]   16 KB (one region at a time)

    const int tid  = threadIdx.x;
    const int lane = tid & 31;
    const int wave = tid >> 5;               // 0..7
    const int grp  = lane >> 4;              // lane group (0: lanes 0-15, 1: 16-31)
    const int lm   = lane & 15;

    // Stage A: pooled[64][103] -> f16, zero-padded to K=128
    for (int idx = tid; idx < BATCH * KPAD; idx += 256) {
        const int b = idx >> 7, k = idx & 127;
        Ah[idx] = (k < C_IN) ? (_Float16)pooled[b * C_IN + k] : (_Float16)0.f;
    }

    for (int r = 0; r < NREG; ++r) {
        __syncthreads();
        // Stage B for region r: Bh[k][n] = W1[r][n][k], zero-padded
        for (int idx = tid; idx < KPAD * HIDDEN; idx += 256) {
            const int k = idx >> 6, n = idx & 63;
            Bh[idx] = (k < C_IN) ? (_Float16)W1[((r * HIDDEN) + n) * C_IN + k]
                                 : (_Float16)0.f;
        }
        __syncthreads();

        // 4x4 output tiles of 16x16, two tiles per wave (uniform -> EXEC all 1s)
        for (int t = wave; t < 16; t += 8) {
            const int mt = t >> 2, nt = t & 3;

            // init accumulator with bias (same for all 8 rows of this lane)
            v8f acc;
            const float bv = b1[r * HIDDEN + nt * 16 + lm];
            #pragma unroll
            for (int j = 0; j < 8; ++j) acc[j] = bv;

            #pragma unroll
            for (int ks = 0; ks < KPAD / 32; ++ks) {
                // 16-bit A layout: lane holds row M=lm; K packed 2/VGPR,
                // lanes 16-31 offset by +8 within each 16-wide K half.
                v16h a, bm;
                #pragma unroll
                for (int e = 0; e < 16; ++e) {
                    const int v = e >> 1, pbit = e & 1;
                    const int kk = ks * 32 + grp * 8 +
                                   ((v < 4) ? (v * 2 + pbit)
                                            : (16 + (v - 4) * 2 + pbit));
                    a[e]  = Ah[(mt * 16 + lm) * KPAD + kk];      // A[M][K]
                    bm[e] = Bh[kk * HIDDEN + nt * 16 + lm];      // B[K][N]
                }
                // D = A x B + C   (f32 accumulate)
                acc = __builtin_amdgcn_wmma_f32_16x16x32_f16(
                        /*neg_a=*/false, a, /*neg_b=*/false, bm,
                        /*c_mod=*/(short)0, acc,
                        /*reuse_a=*/false, /*reuse_b=*/false);
            }

            // C/D layout: VGPR j, lanes 0-15 -> M=j, lanes 16-31 -> M=8+j
            #pragma unroll
            for (int j = 0; j < 8; ++j) {
                const int brow = mt * 16 + grp * 8 + j;
                const int n    = nt * 16 + lm;
                const float v  = acc[j];
                hbuf[(brow * NREG + r) * HIDDEN + n] = v > 0.f ? v : 0.f;
            }
        }
    }
}

// ---------------------------------------------------------------------------
// Kernel 3: layer 2 (K=64, N=3 experts) + top-2 + softmax.
// One thread per (b, r) row.  Outputs concatenated:
//   [0,384)    top_k_indices  [64][3][2]  (as float)
//   [384,768)  top_k_scores   [64][3][2]
//   [768,1344) gate_logits    [64][3][3]
// ---------------------------------------------------------------------------
__global__ __launch_bounds__(256) void gate2_topk(const float* __restrict__ hbuf,
                                                  const float* __restrict__ W2,
                                                  const float* __restrict__ b2,
                                                  float* __restrict__ out) {
    const int idx = threadIdx.x;             // (b*3 + r)
    if (idx >= BATCH * NREG) return;
    const int r = idx % NREG;
    const float* hp = hbuf + idx * HIDDEN;

    float g[NEXP];
    #pragma unroll
    for (int e = 0; e < NEXP; ++e) {
        float acc = b2[r * NEXP + e];
        const float* wp = W2 + ((r * NEXP) + e) * HIDDEN;
        #pragma unroll 8
        for (int k = 0; k < HIDDEN; ++k) acc += hp[k] * wp[k];
        g[e] = acc;
    }

    // top-2 over 3 experts (ties -> lower index first, matching lax.top_k)
    int i0 = 0;
    if (g[1] > g[i0]) i0 = 1;
    if (g[2] > g[i0]) i0 = 2;
    int i1 = -1; float v1 = 0.f;
    #pragma unroll
    for (int e = 0; e < NEXP; ++e) {
        if (e == i0) continue;
        if (i1 < 0 || g[e] > v1) { i1 = e; v1 = g[e]; }
    }
    const float v0  = g[i0];
    const float e1  = __expf(v1 - v0);       // v0 is the max -> stable
    const float inv = 1.0f / (1.0f + e1);

    float* out_idx   = out;                       // [B][R][2]
    float* out_score = out + BATCH * NREG * 2;    // [B][R][2]
    float* out_log   = out + BATCH * NREG * 4;    // [B][R][3]
    out_idx[idx * 2 + 0]   = (float)i0;
    out_idx[idx * 2 + 1]   = (float)i1;
    out_score[idx * 2 + 0] = inv;
    out_score[idx * 2 + 1] = e1 * inv;
    out_log[idx * 3 + 0]   = g[0];
    out_log[idx * 3 + 1]   = g[1];
    out_log[idx * 3 + 2]   = g[2];
}

// ---------------------------------------------------------------------------
extern "C" void kernel_launch(void* const* d_in, const int* in_sizes, int n_in,
                              void* d_out, int out_size, void* d_ws, size_t ws_size,
                              hipStream_t stream) {
    const float* x  = (const float*)d_in[0];   // [64,103,256,256]
    const float* W1 = (const float*)d_in[1];   // [3,64,103]
    const float* b1 = (const float*)d_in[2];   // [3,64]
    const float* W2 = (const float*)d_in[3];   // [3,3,64]
    const float* b2 = (const float*)d_in[4];   // [3,3]
    float* out = (float*)d_out;

    float* pooled = (float*)d_ws;              // 64*103 floats (26368 B, 16B aligned)
    float* hbuf   = pooled + BATCH * C_IN;     // 64*3*64 floats

    pool_kernel<<<BATCH * C_IN, 256, 0, stream>>>(x, pooled);
    gate1_wmma<<<1, 256, 0, stream>>>(pooled, W1, b1, hbuf);
    gate2_topk<<<1, 256, 0, stream>>>(hbuf, W2, b2, out);
}
